// ALiBiAttention_5592047419788
// MI455X (gfx1250) — compile-verified, matrix operand order checked by reading, not run on hardware
//
#include <hip/hip_runtime.h>
#include <hip/hip_bf16.h>

// ---------------------------------------------------------------------------
// ALiBi multi-head attention for MI455X (gfx1250), wave32 + WMMA f16
// + async global->LDS staging (ASYNCcnt) + TDM tensor_load_to_lds (TENSORcnt).
// B=2, T=2048, D=1024, H=16, hd=64.
// ---------------------------------------------------------------------------

#define DMODEL   1024
#define NHEADS   16
#define HDIM     64
#define BATCH    2
#define SEQ      2048

typedef _Float16 half16 __attribute__((ext_vector_type(16)));
typedef float    float8 __attribute__((ext_vector_type(8)));
typedef unsigned int uint4v __attribute__((ext_vector_type(4)));
typedef int      int8v  __attribute__((ext_vector_type(8)));
typedef int      int4v  __attribute__((ext_vector_type(4)));

#if defined(__has_builtin)
#if __has_builtin(__builtin_amdgcn_tensor_load_to_lds)
#define HAVE_TDM 1
#endif
#endif

__device__ __forceinline__ float8 zero8() {
  float8 z = {0.f, 0.f, 0.f, 0.f, 0.f, 0.f, 0.f, 0.f};
  return z;
}

__device__ __forceinline__ float8 wmma_f16(half16 a, half16 b, float8 c) {
  // D = A(16x32 f16) * B(32x16 f16) + C(16x16 f32)
  return __builtin_amdgcn_wmma_f32_16x16x32_f16(
      /*neg_a=*/false, a, /*neg_b=*/false, b,
      /*c_mod=*/(short)0, c, /*reuse_a=*/false, /*reuse_b=*/false);
}

// Async 16-byte global -> LDS copy (per lane), tracked by ASYNCcnt.
__device__ __forceinline__ void async_ld_b128(unsigned lds_off, const void* gptr) {
  asm volatile("global_load_async_to_lds_b128 %0, %1, off"
               :: "v"(lds_off), "v"((unsigned long long)(size_t)gptr)
               : "memory");
}
__device__ __forceinline__ void wait_async0() {
  asm volatile("s_wait_asynccnt 0x0" ::: "memory");
}

#ifdef HAVE_TDM
// TDM: DMA a 2D tile (tile_d1 rows x tile_d0 elements, f16) from global into
// LDS. Descriptor per CDNA5 ISA ch.8 (D# groups 0/1; groups 2/3 zero for 2D).
__device__ __forceinline__ void tdm_load_2d_f16(unsigned lds_off, const void* gaddr,
                                                unsigned tile_d0, unsigned tile_d1,
                                                unsigned tensor_d0, unsigned tensor_d1,
                                                unsigned stride0_elems) {
  unsigned long long ga = (unsigned long long)(size_t)gaddr;
  uint4v g0;
  g0.x = 1u;                                   // count=1 descriptor, load, no gather
  g0.y = lds_off;                              // lds_addr (bytes)
  g0.z = (unsigned)ga;                         // global_addr[31:0]
  g0.w = (unsigned)((ga >> 32) & 0x01FFFFFFu)  // global_addr[56:32]
         | (2u << 30);                         // type = 2 ("image")
  int8v g1;
  g1[0] = (int)(1u << 16);                                        // data_size=1 -> 2B
  g1[1] = (int)((tensor_d0 & 0xFFFFu) << 16);                     // tensor_dim0[15:0]
  g1[2] = (int)((tensor_d0 >> 16) | ((tensor_d1 & 0xFFFFu) << 16));
  g1[3] = (int)((tensor_d1 >> 16) | (tile_d0 << 16));             // tile_dim0
  g1[4] = (int)tile_d1;                                           // tile_dim1, tile_dim2=0
  g1[5] = (int)stride0_elems;                                     // dim0_stride[31:0]
  g1[6] = 0;
  g1[7] = 0;
  int4v z4 = {0, 0, 0, 0};
#if __clang_major__ >= 23
  int8v z8 = {0, 0, 0, 0, 0, 0, 0, 0};
  __builtin_amdgcn_tensor_load_to_lds(g0, g1, z4, z4, z8, 0);
#else
  __builtin_amdgcn_tensor_load_to_lds(g0, g1, z4, z4, 0);
#endif
}
__device__ __forceinline__ void wait_tensor0() {
#if defined(__has_builtin) && __has_builtin(__builtin_amdgcn_s_wait_tensorcnt)
  __builtin_amdgcn_s_wait_tensorcnt((short)0);
#else
  asm volatile("s_wait_tensorcnt 0x0" ::: "memory");
#endif
}
#endif  // HAVE_TDM

// Fragment loader for the CDNA5 16-bit A layout (and the symmetric B layout
// when the source is stored [N,K] row-major):
//   lanes 0-15 hold row (M or N) = lane, hi half selects K-subgroups:
//   hi=0 -> K {0..7, 16..23}, hi=1 -> K {8..15, 24..31}; pairs contiguous,
//   so each fragment is two 16-byte loads per lane.
__device__ __forceinline__ half16 load_frag(const _Float16* __restrict__ base,
                                            int ld, int row, int k0, int hi) {
  const _Float16* p = base + (size_t)row * ld + k0;
  half16 f;
#pragma unroll
  for (int j = 0; j < 8; ++j) {
    int k = (j < 4) ? (8 * hi + 2 * j) : (16 + 8 * hi + 2 * (j - 4));
    f[2 * j]     = p[k];
    f[2 * j + 1] = p[k + 1];
  }
  return f;
}

// ---------------------------------------------------------------------------
// Conversion kernels
// ---------------------------------------------------------------------------
__global__ void alibi_cvt_f16(const float* __restrict__ src,
                              _Float16* __restrict__ dst, int n) {
  int i = blockIdx.x * blockDim.x + threadIdx.x;
  if (i < n) dst[i] = (_Float16)src[i];
}

// dst[n][k] = (f16) src[k][n]  for a DMODEL x DMODEL weight
__global__ void alibi_cvt_wT(const float* __restrict__ src,
                             _Float16* __restrict__ dst) {
  int i = blockIdx.x * blockDim.x + threadIdx.x;  // over D*D
  int n = i >> 10;          // /1024
  int k = i & (DMODEL - 1); // %1024
  dst[i] = (_Float16)src[k * DMODEL + n];
}

// ---------------------------------------------------------------------------
// WMMA GEMM:  C[M,N] = A[M,K] (f16, row-major) * Bt[N,K] (f16, row-major) + bias
// One wave computes a 16x64 tile of C (4 accumulators). Block = 8 waves
// stacked along M -> 128x64 block tile. The shared 64x32 Bt tile is staged
// into LDS once per K-step via async global->LDS copies (1 b128 per thread).
//   mode 0: write f16, head-major   out[b][h][t][d]
//   mode 1: write f16, head-major transposed (for V)  out[b][h][d][t]
//   mode 2: write f32, row-major [M,N] (final output)
// ---------------------------------------------------------------------------
__global__ __launch_bounds__(256)
void alibi_gemm16(const _Float16* __restrict__ A,
                  const _Float16* __restrict__ Bt,
                  const float* __restrict__ bias,
                  void* __restrict__ out,
                  int M, int N, int K, int T, int mode) {
  __shared__ _Float16 bstage[64][32];  // 4 KB: Bt rows n0..n0+63, k-slice of 32

  const int tid  = threadIdx.x;
  const int wave = tid >> 5;
  const int lane = tid & 31;
  const int lh = lane & 15;
  const int hi = lane >> 4;

  const int m0 = (blockIdx.x * 8 + wave) * 16;
  const int n0 = blockIdx.y * 64;

  // cooperative staging assignment: 64 rows x 64B, 16B per thread
  const int srow = tid >> 2;   // 0..63
  const int sch  = tid & 3;    // 0..3 (8 halfs each)
  const unsigned lds_off = (unsigned)(size_t)&bstage[srow][sch * 8];
  const _Float16* gsrc_row = Bt + (size_t)(n0 + srow) * K + sch * 8;

  float8 acc0 = zero8(), acc1 = zero8(), acc2 = zero8(), acc3 = zero8();

  const _Float16* arow = A + (size_t)(m0 + lh) * K;
  for (int k0 = 0; k0 < K; k0 += 32) {
    // stage the shared Bt tile into LDS (async engine)
    async_ld_b128(lds_off, gsrc_row + k0);
    if (k0 + 32 < K) __builtin_prefetch(arow + k0 + 32, 0, 3);
    wait_async0();
    __syncthreads();

    half16 a  = load_frag(A, K, m0 + lh, k0, hi);
    half16 b0 = load_frag(&bstage[0][0], 32, 0  + lh, 0, hi);
    half16 b1 = load_frag(&bstage[0][0], 32, 16 + lh, 0, hi);
    half16 b2 = load_frag(&bstage[0][0], 32, 32 + lh, 0, hi);
    half16 b3 = load_frag(&bstage[0][0], 32, 48 + lh, 0, hi);
    acc0 = wmma_f16(a, b0, acc0);
    acc1 = wmma_f16(a, b1, acc1);
    acc2 = wmma_f16(a, b2, acc2);
    acc3 = wmma_f16(a, b3, acc3);
    __syncthreads();  // protect bstage before next iteration overwrites
  }

  float8 accs[4] = {acc0, acc1, acc2, acc3};
#pragma unroll
  for (int t = 0; t < 4; ++t) {
#pragma unroll
    for (int r = 0; r < 8; ++r) {
      int m = m0 + 8 * hi + r;     // C row (M = 8*hi + r mapping)
      int n = n0 + 16 * t + lh;    // C col (N = lane)
      float v = accs[t][r] + bias[n];
      if (mode == 2) {
        ((float*)out)[(size_t)m * N + n] = v;
      } else {
        int bb = m / T, tt = m - (m / T) * T;
        int hh = n >> 6, dd = n & (HDIM - 1);
        _Float16* oh = (_Float16*)out;
        if (mode == 0)
          oh[(((size_t)bb * NHEADS + hh) * T + tt) * HDIM + dd] = (_Float16)v;
        else
          oh[(((size_t)bb * NHEADS + hh) * HDIM + dd) * T + tt] = (_Float16)v;
      }
    }
  }
}

// ---------------------------------------------------------------------------
// Flash-style ALiBi attention.
// One block = 8 waves sharing (b,h); each wave owns a 16-query tile.
// The 32-key K chunk is DMA'd into LDS by the Tensor Data Mover (wave 0
// issues one tensor_load_to_lds per chunk); the V chunk uses per-lane async
// global->LDS copies.
// Q: [b][h][T][hd] f16, K: [b][h][T][hd] f16, Vt: [b][h][hd][T] f16.
// Oh: f16 [B*T, DMODEL] row-major (A matrix for the final projection).
// ---------------------------------------------------------------------------
__global__ __launch_bounds__(256)
void alibi_attn(const _Float16* __restrict__ Qh,
                const _Float16* __restrict__ Kh,
                const _Float16* __restrict__ Vt,
                _Float16* __restrict__ Oh) {
  __shared__ _Float16 kst[32][64];      // 4 KB: keys kt..kt+31, hd 0..63
  __shared__ _Float16 vst[64][32];      // 4 KB: hd rows, keys kt..kt+31
  __shared__ _Float16 plds[8][16][36];  // per-wave P tile staging (padded)

  const int tid  = threadIdx.x;
  const int wave = tid >> 5;
  const int lane = tid & 31;
  const int lh = lane & 15;
  const int hi = lane >> 4;

  const int qpb  = SEQ / 128;                 // query-blocks per (b,h)
  const int qblk = blockIdx.x % qpb;
  const int bh   = blockIdx.x / qpb;
  const int h = bh % NHEADS;
  const int b = bh / NHEADS;
  const int q0 = qblk * 128 + wave * 16;

  const _Float16* Qp = Qh + (size_t)bh * SEQ * HDIM;   // [T][hd]
  const _Float16* Kp = Kh + (size_t)bh * SEQ * HDIM;   // [T][hd]
  const _Float16* Vp = Vt + (size_t)bh * HDIM * SEQ;   // [hd][T]

  const float slope = exp2f(-0.5f * (float)(h + 1));   // ALiBi slope, H=16
  const float scale = 0.125f;                          // hd^-0.5

  // V staging assignment (16B per thread)
  const int vrow = tid >> 2, vch = tid & 3;   // V: 64 rows x 64B
  const unsigned voff = (unsigned)(size_t)&vst[vrow][vch * 8];
  const _Float16* vg_row = Vp + (size_t)vrow * SEQ + vch * 8;

#ifdef HAVE_TDM
  const unsigned kst_base = (unsigned)(size_t)&kst[0][0];
#else
  const int krow = tid >> 3, kch = tid & 7;   // K: 32 rows x 128B
  const unsigned koff = (unsigned)(size_t)&kst[krow][kch * 8];
  const _Float16* kg_row = Kp + (size_t)krow * HDIM + kch * 8;
#endif

  // Q fragments (reused for every key chunk): rows q0..q0+15, K-dim = hd
  half16 aq0 = load_frag(Qp, HDIM, q0 + lh, 0,  hi);
  half16 aq1 = load_frag(Qp, HDIM, q0 + lh, 32, hi);

  float8 o0 = zero8(), o1 = zero8(), o2 = zero8(), o3 = zero8();
  float m_i[8], l_i[8];
#pragma unroll
  for (int r = 0; r < 8; ++r) { m_i[r] = -INFINITY; l_i[r] = 0.f; }

#pragma unroll 1
  for (int kt = 0; kt < SEQ; kt += 32) {
    // ---- stage K chunk (32x64) via TDM, V chunk (64x32) via async ----
#ifdef HAVE_TDM
    if (wave == 0) {
      tdm_load_2d_f16(kst_base, Kp + (size_t)kt * HDIM,
                      /*tile_d0=*/HDIM, /*tile_d1=*/32,
                      /*tensor_d0=*/HDIM, /*tensor_d1=*/(unsigned)(SEQ - kt),
                      /*stride0=*/HDIM);
      wait_tensor0();
    }
#else
    async_ld_b128(koff, kg_row + (size_t)kt * HDIM);
#endif
    async_ld_b128(voff, vg_row + kt);
    wait_async0();
    __syncthreads();

    // ---- S = (Q K^T) for 32 keys: two 16x16 tiles, contraction hd=64 ----
    float8 s0 = zero8(), s1 = zero8();
    {
      half16 bk = load_frag(&kst[0][0], HDIM, lh, 0, hi);
      s0 = wmma_f16(aq0, bk, s0);
      bk = load_frag(&kst[0][0], HDIM, lh, 32, hi);
      s0 = wmma_f16(aq1, bk, s0);
      bk = load_frag(&kst[0][0], HDIM, 16 + lh, 0, hi);
      s1 = wmma_f16(aq0, bk, s1);
      bk = load_frag(&kst[0][0], HDIM, 16 + lh, 32, hi);
      s1 = wmma_f16(aq1, bk, s1);
    }

    // ---- online softmax with fused ALiBi bias ----
    float p0[8], p1[8], alpha[8];
#pragma unroll
    for (int r = 0; r < 8; ++r) {
      int q = q0 + 8 * hi + r;
      float v0 = s0[r] * scale - slope * fabsf((float)(q - (kt + lh)));
      float v1 = s1[r] * scale - slope * fabsf((float)(q - (kt + 16 + lh)));
      float mc = fmaxf(v0, v1);
      mc = fmaxf(mc, __shfl_xor(mc, 1, 32));
      mc = fmaxf(mc, __shfl_xor(mc, 2, 32));
      mc = fmaxf(mc, __shfl_xor(mc, 4, 32));
      mc = fmaxf(mc, __shfl_xor(mc, 8, 32));
      float mn = fmaxf(m_i[r], mc);
      float al = __expf(m_i[r] - mn);
      float e0 = __expf(v0 - mn);
      float e1 = __expf(v1 - mn);
      float rs = e0 + e1;
      rs += __shfl_xor(rs, 1, 32);
      rs += __shfl_xor(rs, 2, 32);
      rs += __shfl_xor(rs, 4, 32);
      rs += __shfl_xor(rs, 8, 32);
      l_i[r] = l_i[r] * al + rs;
      m_i[r] = mn;
      alpha[r] = al;
      p0[r] = e0;
      p1[r] = e1;
    }
#pragma unroll
    for (int r = 0; r < 8; ++r) {
      o0[r] *= alpha[r];
      o1[r] *= alpha[r];
      o2[r] *= alpha[r];
      o3[r] *= alpha[r];
    }

    // ---- re-layout P (C layout -> A layout) through LDS ----
#pragma unroll
    for (int r = 0; r < 8; ++r) {
      plds[wave][8 * hi + r][lh]      = (_Float16)p0[r];
      plds[wave][8 * hi + r][16 + lh] = (_Float16)p1[r];
    }
    __asm__ volatile("s_wait_dscnt 0" ::: "memory");
    half16 pa = load_frag(&plds[wave][0][0], 36, lh, 0, hi);
    __asm__ volatile("" ::: "memory");

    // ---- O += P * V (contraction over 32 keys), hd in 4 N-tiles ----
    o0 = wmma_f16(pa, load_frag(&vst[0][0], 32, 0  + lh, 0, hi), o0);
    o1 = wmma_f16(pa, load_frag(&vst[0][0], 32, 16 + lh, 0, hi), o1);
    o2 = wmma_f16(pa, load_frag(&vst[0][0], 32, 32 + lh, 0, hi), o2);
    o3 = wmma_f16(pa, load_frag(&vst[0][0], 32, 48 + lh, 0, hi), o3);

    __syncthreads();  // protect kst/vst before next chunk overwrites
  }

  // ---- normalize and write O as f16 [B*T, DMODEL] row-major ----
  float8 os[4] = {o0, o1, o2, o3};
#pragma unroll
  for (int t = 0; t < 4; ++t) {
#pragma unroll
    for (int r = 0; r < 8; ++r) {
      int q = q0 + 8 * hi + r;
      float inv = 1.0f / l_i[r];
      int col = h * HDIM + 16 * t + lh;
      Oh[((size_t)b * SEQ + q) * DMODEL + col] = (_Float16)(os[t][r] * inv);
    }
  }
}

// ---------------------------------------------------------------------------
// Launch
// ---------------------------------------------------------------------------
extern "C" void kernel_launch(void* const* d_in, const int* in_sizes, int n_in,
                              void* d_out, int out_size, void* d_ws, size_t ws_size,
                              hipStream_t stream) {
  const float* query = (const float*)d_in[0];
  const float* key   = (const float*)d_in[1];
  const float* value = (const float*)d_in[2];
  const float* Wq    = (const float*)d_in[3];
  const float* bq    = (const float*)d_in[4];
  const float* Wk    = (const float*)d_in[5];
  const float* bk    = (const float*)d_in[6];
  const float* Wv    = (const float*)d_in[7];
  const float* bv    = (const float*)d_in[8];
  const float* Wo    = (const float*)d_in[9];
  const float* bo    = (const float*)d_in[10];

  const size_t M  = (size_t)BATCH * SEQ;   // 4096
  const size_t XE = M * DMODEL;            // 4M elements
  const size_t WE = (size_t)DMODEL * DMODEL;

  _Float16* Xq  = (_Float16*)d_ws;
  _Float16* Xk  = Xq  + XE;
  _Float16* Xv  = Xk  + XE;
  _Float16* WqT = Xv  + XE;
  _Float16* WkT = WqT + WE;
  _Float16* WvT = WkT + WE;
  _Float16* WoT = WvT + WE;
  _Float16* Qh  = WoT + WE;
  _Float16* Kh  = Qh  + XE;
  _Float16* Vt  = Kh  + XE;
  _Float16* Oh  = Vt  + XE;

  dim3 blk(256);

  // 1) convert activations to f16
  alibi_cvt_f16<<<(int)(XE / 256), blk, 0, stream>>>(query, Xq, (int)XE);
  alibi_cvt_f16<<<(int)(XE / 256), blk, 0, stream>>>(key,   Xk, (int)XE);
  alibi_cvt_f16<<<(int)(XE / 256), blk, 0, stream>>>(value, Xv, (int)XE);

  // 2) convert+transpose weights
  alibi_cvt_wT<<<(int)(WE / 256), blk, 0, stream>>>(Wq, WqT);
  alibi_cvt_wT<<<(int)(WE / 256), blk, 0, stream>>>(Wk, WkT);
  alibi_cvt_wT<<<(int)(WE / 256), blk, 0, stream>>>(Wv, WvT);
  alibi_cvt_wT<<<(int)(WE / 256), blk, 0, stream>>>(Wo, WoT);

  // 3) Q/K/V projections (WMMA)
  dim3 gg((unsigned)(M / 128), DMODEL / 64);
  alibi_gemm16<<<gg, blk, 0, stream>>>(Xq, WqT, bq, Qh, (int)M, DMODEL, DMODEL, SEQ, 0);
  alibi_gemm16<<<gg, blk, 0, stream>>>(Xk, WkT, bk, Kh, (int)M, DMODEL, DMODEL, SEQ, 0);
  alibi_gemm16<<<gg, blk, 0, stream>>>(Xv, WvT, bv, Vt, (int)M, DMODEL, DMODEL, SEQ, 1);

  // 4) flash-style ALiBi attention (WMMA + TDM)
  alibi_attn<<<BATCH * NHEADS * (SEQ / 128), blk, 0, stream>>>(Qh, Kh, Vt, Oh);

  // 5) output projection -> f32 d_out (WMMA)
  alibi_gemm16<<<gg, blk, 0, stream>>>(Oh, WoT, bo, d_out, (int)M, DMODEL, DMODEL, SEQ, 2);
}
